// WhisperAttention_19593640804376
// MI455X (gfx1250) — matrix operand order repaired and checked by reading, not compiled, not run on hardware
//
#include <hip/hip_runtime.h>
#include <hip/hip_bf16.h>

typedef __attribute__((ext_vector_type(16))) _Float16 v16h;
typedef __attribute__((ext_vector_type(8)))  _Float16 v8h;
typedef __attribute__((ext_vector_type(8)))  float    v8f;

#define DD    1280
#define NH    20
#define HDIM  64
#define NB    4
#define ENC   1500
#define QTOT  1600
#define KVTOT 6000
#define ENCP  1504   // padded per-request kv stride (multiple of 16 -> 32B aligned)
#define KVP   (NB * ENCP)

__device__ __forceinline__ v8f wmma_f16(v16h a, v16h b, v8f c) {
  return __builtin_amdgcn_wmma_f32_16x16x32_f16(false, a, false, b, (short)0, c,
                                                false, false);
}

// Build A-operand fragment (16x32 f16): lane L holds row L%16,
// elements 0..15 = K (L/16)*16 .. +15 (contiguous)
__device__ __forceinline__ v16h make_a(const _Float16* p0, const _Float16* p1) {
  v8h lo = *(const v8h*)p0;
  v8h hi = *(const v8h*)p1;
  v16h a;
#pragma unroll
  for (int i = 0; i < 8; ++i) { a[i] = lo[i]; a[8 + i] = hi[i]; }
  return a;
}

// ---------------------------------------------------------------- converts
__global__ __launch_bounds__(256) void cvt_f16(const float* __restrict__ src,
                                               _Float16* __restrict__ dst, int n) {
  int i = blockIdx.x * 256 + threadIdx.x;
  if (i < n) dst[i] = (_Float16)src[i];
}

// src: K x N row-major (f32); dst: N x K row-major f16 (i.e. transposed)
__global__ __launch_bounds__(256) void cvt_tr(const float* __restrict__ src,
                                              _Float16* __restrict__ dst,
                                              int K, int N) {
  int k = blockIdx.x * 256 + threadIdx.x;
  int n = blockIdx.y;
  if (k < K) dst[(size_t)n * K + k] = (_Float16)src[(size_t)k * N + n];
}

// ---------------------------------------------------------------- GEMM
// C(MxN) = A(MxK) @ B(KxN) + bias, with B supplied transposed (BT: N x K).
// 4x4 register blocking: each wave owns a 64x64 tile (16 WMMA accumulators),
// 4 waves in 2x2 -> 128x128 block tile. N must divide 128; M edge-clamped.
// mode 0: q-proj  -> out16 f16 row-major, scaled by 1/8
// mode 1: kv-proj -> n<DD: k16 row-major; n>=DD: vT padded [DD][KVP]
// mode 2: o-proj  -> outf f32 row-major
__global__ __launch_bounds__(128) void gemm16(const _Float16* __restrict__ A,
                                              const _Float16* __restrict__ BT,
                                              const float* __restrict__ bias,
                                              int M, int K, int mode,
                                              _Float16* __restrict__ out16,
                                              _Float16* __restrict__ out16b,
                                              float* __restrict__ outf) {
  const int wid   = threadIdx.x >> 5;
  const int lane  = threadIdx.x & 31;
  const int l16   = lane & 15;
  const int khalf = lane >> 4;
  const int m0 = blockIdx.x * 128 + (wid >> 1) * 64;
  const int n0 = blockIdx.y * 128 + (wid & 1) * 64;

  const _Float16* arow[4];
  const _Float16* brow[4];
#pragma unroll
  for (int i = 0; i < 4; ++i) {
    int r = m0 + i * 16 + l16;
    if (r > M - 1) r = M - 1;                 // clamp; guarded at store
    arow[i] = A  + (size_t)r * K;
    brow[i] = BT + (size_t)(n0 + i * 16 + l16) * K;
  }

  v8f acc[4][4];
#pragma unroll
  for (int mi = 0; mi < 4; ++mi)
#pragma unroll
    for (int ni = 0; ni < 4; ++ni) acc[mi][ni] = (v8f){};

  for (int kb = 0; kb < K; kb += 32) {
    __builtin_prefetch((const void*)(brow[0] + kb + 256), 0, 1);
    v16h af[4], bf[4];
#pragma unroll
    for (int i = 0; i < 4; ++i) {
      af[i] = make_a(arow[i] + kb + khalf * 16, arow[i] + kb + 8 + khalf * 16);
      bf[i] = *(const v16h*)(brow[i] + kb + khalf * 16);
    }
#pragma unroll
    for (int mi = 0; mi < 4; ++mi)
#pragma unroll
      for (int ni = 0; ni < 4; ++ni)
        acc[mi][ni] = wmma_f16(af[mi], bf[ni], acc[mi][ni]);
  }

#pragma unroll
  for (int ni = 0; ni < 4; ++ni) {
    const int n  = n0 + ni * 16 + l16;
    const float bv = bias[n];
#pragma unroll
    for (int mi = 0; mi < 4; ++mi) {
#pragma unroll
      for (int r = 0; r < 8; ++r) {
        const int row = m0 + mi * 16 + r + (khalf << 3); // C layout rows
        if (row < M) {
          const float v = acc[mi][ni][r] + bv;
          if (mode == 0) {
            out16[(size_t)row * DD + n] = (_Float16)(v * 0.125f);
          } else if (mode == 1) {
            if (n < DD) {
              out16[(size_t)row * DD + n] = (_Float16)v;
            } else {
              const int d  = n - DD;
              const int bb = row / ENC;
              const int e  = row - bb * ENC;
              out16b[(size_t)d * KVP + bb * ENCP + e] = (_Float16)v;
            }
          } else {
            outf[(size_t)row * DD + n] = v;
          }
        }
      }
    }
  }
}

// ---------------------------------------------------------------- attention
// One wave per (head, 16-query tile within one request). Flash-style online
// softmax over the request's 1500 kv in 32-wide chunks.
__global__ __launch_bounds__(32) void attn_kernel(const _Float16* __restrict__ q16,
                                                  const _Float16* __restrict__ k16,
                                                  const _Float16* __restrict__ vT,
                                                  const int* __restrict__ seq_lens,
                                                  _Float16* __restrict__ attn16) {
  const int h = blockIdx.y;
  const int t = blockIdx.x;

  // map global tile index -> (request b, row offset inside request)
  int b = -1, row_in_req = 0, cumq = 0, len = 0;
  int acc = 0, cum = 0;
  for (int i = 0; i < NB; ++i) {
    const int li = seq_lens[i];
    const int nt = (li + 15) >> 4;
    if (b < 0 && t < acc + nt) { b = i; row_in_req = (t - acc) << 4; cumq = cum; len = li; }
    acc += nt; cum += li;
  }
  if (b < 0) return;                       // block-uniform exit
  const int nvalid = (len - row_in_req < 16) ? (len - row_in_req) : 16;
  const int row0g  = cumq + row_in_req;

  const int lane  = threadIdx.x;
  const int l16   = lane & 15;
  const int khalf = lane >> 4;

  // Q tile as two A fragments covering head-dim K = 0..63 (lane-clamped rows)
  int m = l16; if (m >= nvalid) m = nvalid - 1;
  const _Float16* qrow = q16 + (size_t)(row0g + m) * DD + h * HDIM;
  v16h aq0 = make_a(qrow + 0  + khalf * 16, qrow + 8  + khalf * 16);
  v16h aq1 = make_a(qrow + 32 + khalf * 16, qrow + 40 + khalf * 16);

  float mrow[8], lrow[8];
  v8f o[4];
#pragma unroll
  for (int r = 0; r < 8; ++r) { mrow[r] = -INFINITY; lrow[r] = 0.f; }
#pragma unroll
  for (int nt = 0; nt < 4; ++nt) o[nt] = (v8f){};

  __shared__ _Float16 ldsP[16 * 32];
  const size_t kvbase  = (size_t)b * ENC;
  const size_t kvbaseP = (size_t)b * ENCP;

  for (int kc = 0; kc < ENC; kc += 32) {
    // ---- scores: two 16x16 tiles (K = head dim, 2 x K32 WMMA each)
    v8f s[2];
#pragma unroll
    for (int nh2 = 0; nh2 < 2; ++nh2) {
      const int col = kc + nh2 * 16 + l16;            // kv column (this lane)
      const int kvr = (col < ENC) ? col : (ENC - 1);  // clamp, mask below
      const _Float16* krow = k16 + (kvbase + kvr) * DD + h * HDIM;
      v16h bk0 = *(const v16h*)(krow + 0  + khalf * 16);
      v16h bk1 = *(const v16h*)(krow + 32 + khalf * 16);
      v8f c = {};
      c = wmma_f16(aq0, bk0, c);
      c = wmma_f16(aq1, bk1, c);
      if (col >= ENC) {
#pragma unroll
        for (int r = 0; r < 8; ++r) c[r] = -INFINITY;
      }
      s[nh2] = c;
    }

    // ---- online softmax (rows live across 16-lane halves; xor 1/2/4/8 reduces)
    float newm[8];
#pragma unroll
    for (int r = 0; r < 8; ++r) {
      float mx = fmaxf(s[0][r], s[1][r]);
      mx = fmaxf(mx, __shfl_xor(mx, 1, 32));
      mx = fmaxf(mx, __shfl_xor(mx, 2, 32));
      mx = fmaxf(mx, __shfl_xor(mx, 4, 32));
      mx = fmaxf(mx, __shfl_xor(mx, 8, 32));
      newm[r] = fmaxf(mrow[r], mx);
    }
#pragma unroll
    for (int r = 0; r < 8; ++r) {
      const float alpha = expf(mrow[r] - newm[r]);
      mrow[r] = newm[r];
      const float p0 = expf(s[0][r] - newm[r]);
      const float p1 = expf(s[1][r] - newm[r]);
      float sum = p0 + p1;
      sum += __shfl_xor(sum, 1, 32);
      sum += __shfl_xor(sum, 2, 32);
      sum += __shfl_xor(sum, 4, 32);
      sum += __shfl_xor(sum, 8, 32);
      lrow[r] = lrow[r] * alpha + sum;
      const int mm = r + (khalf << 3);
      ldsP[mm * 32 + l16]      = (_Float16)p0;
      ldsP[mm * 32 + 16 + l16] = (_Float16)p1;
#pragma unroll
      for (int nt = 0; nt < 4; ++nt) o[nt][r] *= alpha;
    }
    __syncthreads();

    // re-stage P (C layout -> A layout) via LDS
    const _Float16* pr = &ldsP[l16 * 32 + khalf * 16];
    v16h ap = make_a(pr, pr + 8);

    // ---- PV: accumulate straight into WMMA C operand
#pragma unroll
    for (int nt = 0; nt < 4; ++nt) {
      const int d = h * HDIM + nt * 16 + l16;
      v16h bv = *(const v16h*)(vT + (size_t)d * KVP + kvbaseP + kc + khalf * 16);
      o[nt] = wmma_f16(ap, bv, o[nt]);
    }
    __syncthreads();
  }

  // ---- normalize + store valid rows
#pragma unroll
  for (int r = 0; r < 8; ++r) {
    const int mm = r + (khalf << 3);
    if (mm < nvalid) {
      const float inv = 1.f / lrow[r];
      const size_t ro = (size_t)(row0g + mm) * DD + h * HDIM;
#pragma unroll
      for (int nt = 0; nt < 4; ++nt)
        attn16[ro + nt * 16 + l16] = (_Float16)(o[nt][r] * inv);
    }
  }
}

// ---------------------------------------------------------------- launch
extern "C" void kernel_launch(void* const* d_in, const int* in_sizes, int n_in,
                              void* d_out, int out_size, void* d_ws, size_t ws_size,
                              hipStream_t stream) {
  const float* hs  = (const float*)d_in[0];
  const float* xhs = (const float*)d_in[1];
  const int*   sl  = (const int*)d_in[2];
  const float* Wq  = (const float*)d_in[3];
  const float* bq  = (const float*)d_in[4];
  const float* Wkv = (const float*)d_in[5];
  const float* bkv = (const float*)d_in[6];
  const float* Wo  = (const float*)d_in[7];
  const float* bo  = (const float*)d_in[8];
  float* out = (float*)d_out;

  char* ws = (char*)d_ws;
  size_t off = 0;
  auto alloc = [&](size_t bytes) {
    char* p = ws + off;
    off += (bytes + 255) & ~(size_t)255;
    return p;
  };
  _Float16* h16    = (_Float16*)alloc((size_t)QTOT  * DD * 2);
  _Float16* x16    = (_Float16*)alloc((size_t)KVTOT * DD * 2);
  _Float16* wqT    = (_Float16*)alloc((size_t)DD * DD * 2);
  _Float16* wkvT   = (_Float16*)alloc((size_t)2 * DD * DD * 2);
  _Float16* woT    = (_Float16*)alloc((size_t)DD * DD * 2);
  _Float16* q16    = (_Float16*)alloc((size_t)QTOT  * DD * 2);
  _Float16* k16    = (_Float16*)alloc((size_t)KVTOT * DD * 2);
  _Float16* vT     = (_Float16*)alloc((size_t)DD * KVP * 2);
  _Float16* attn16 = (_Float16*)alloc((size_t)QTOT  * DD * 2);

  // 1. conversions / weight transposes
  {
    int n = QTOT * DD;
    cvt_f16<<<(n + 255) / 256, 256, 0, stream>>>(hs, h16, n);
    n = KVTOT * DD;
    cvt_f16<<<(n + 255) / 256, 256, 0, stream>>>(xhs, x16, n);
    cvt_tr<<<dim3((DD + 255) / 256, DD),     256, 0, stream>>>(Wq,  wqT,  DD, DD);
    cvt_tr<<<dim3((DD + 255) / 256, 2 * DD), 256, 0, stream>>>(Wkv, wkvT, DD, 2 * DD);
    cvt_tr<<<dim3((DD + 255) / 256, DD),     256, 0, stream>>>(Wo,  woT,  DD, DD);
  }
  // 2. Q projection (scaled), f16 out   (M=1600 -> 13 row-blocks of 128)
  gemm16<<<dim3((QTOT + 127) / 128, DD / 128), 128, 0, stream>>>(
      h16, wqT, bq, QTOT, DD, 0, q16, nullptr, nullptr);
  // 3. KV projection: K row-major + V transposed (padded per-request stride)
  gemm16<<<dim3((KVTOT + 127) / 128, (2 * DD) / 128), 128, 0, stream>>>(
      x16, wkvT, bkv, KVTOT, DD, 1, k16, vT, nullptr);
  // 4. flash attention per (head, query tile); max tiles = Q/16 + B
  attn_kernel<<<dim3(QTOT / 16 + NB, NH), 32, 0, stream>>>(
      q16, k16, vT, sl, attn16);
  // 5. output projection, f32 straight into d_out
  gemm16<<<dim3((QTOT + 127) / 128, DD / 128), 128, 0, stream>>>(
      attn16, woT, bo, QTOT, DD, 2, nullptr, nullptr, out);
}
